// HybridQLSTM_85873576117068
// MI455X (gfx1250) — matrix operand order repaired and checked by reading, not compiled, not run on hardware
//
#include <hip/hip_runtime.h>
#include <hip/hip_bf16.h>

#define V_  32000
#define E_  256
#define H_  512
#define B_  128
#define S_  1024
#define K0_ (E_ + H_)   /* 768  : layer0 input  */
#define K1_ (2 * H_)    /* 1024 : layer1 input  */
#define NWG 128
#define TPB 256
#define UW  4           /* hidden units per workgroup -> 16 gate rows */

typedef __attribute__((ext_vector_type(16))) __bf16 v16bf;
typedef __attribute__((ext_vector_type(8)))  __bf16 v8bf;
typedef __attribute__((ext_vector_type(8)))  float  v8f;

union ABu { v16bf v; v8bf h[2]; };

__device__ __forceinline__ unsigned short f2bf(float f) {
  unsigned int u = __float_as_uint(f);
  u += 0x7FFFu + ((u >> 16) & 1u);          // round-to-nearest-even
  return (unsigned short)(u >> 16);
}
__device__ __forceinline__ float bf2f(unsigned short s) {
  return __uint_as_float(((unsigned int)s) << 16);
}

#define LOG2E_ 1.4426950408889634f
// Fast sigmoid/tanh on the v_exp_f32 + v_rcp_f32 TRANS pipes (no precise-div expansion).
__device__ __forceinline__ float sigf(float x) {
  float e = __builtin_amdgcn_exp2f(-x * LOG2E_);        // exp(-x) = exp2(-x*log2e)
  return __builtin_amdgcn_rcpf(1.0f + e);
}
__device__ __forceinline__ float tanhfast(float x) {
  float e = __builtin_amdgcn_exp2f(2.0f * LOG2E_ * x);  // exp(2x) = exp2(2x*log2e)
  float r = __builtin_amdgcn_rcpf(1.0f + e);
  return __builtin_fmaf(-2.0f, r, 1.0f);                // 1 - 2/(1+e^{2x})
}

// Grid-wide barrier for a persistent kernel (counter + epoch in global mem).
__device__ __forceinline__ void gsync(int* cnt, int* ep) {
  __syncthreads();
  if (threadIdx.x == 0) {
    __threadfence();
    int e = atomicAdd(ep, 0);               // read epoch BEFORE arriving
    if (atomicAdd(cnt, 1) == NWG - 1) {
      atomicExch(cnt, 0);
      __threadfence();
      atomicAdd(ep, 1);
    } else {
      while (atomicAdd(ep, 0) == e) __builtin_amdgcn_s_sleep(1);
    }
  }
  __syncthreads();
}

__global__ __launch_bounds__(TPB, 1) void lstm_persist(
    const int* __restrict__ x, const float* __restrict__ emb,
    const float* __restrict__ W0, const float* __restrict__ b0, const float* __restrict__ bb0,
    const float* __restrict__ W1, const float* __restrict__ b1, const float* __restrict__ bb1,
    int* sync_cnt, int* sync_ep,
    unsigned short* act0, unsigned short* act1)
{
  // Per-WG weight slices (bf16) resident in LDS for the whole sequence.
  __shared__ __attribute__((aligned(32))) unsigned short w0s[16 * K0_]; // 24 KB
  __shared__ __attribute__((aligned(32))) unsigned short w1s[16 * K1_]; // 32 KB
  __shared__ __attribute__((aligned(16))) float zs[8][16][16];          // 8 KB gate-tile scratch
  __shared__ float bias0s[16], bias1s[16];

  const int tid   = threadIdx.x;
  const int lane  = tid & 31;
  const int wv    = tid >> 5;       // wave 0..7 -> batch tile
  const int wg    = blockIdx.x;     // 0..127   -> hidden-unit slice (and batch row for gather)
  const int u0    = wg * UW;
  const int bbase = wv * 16;
  const int mn    = lane & 15;      // A-row (gate row) / B-col (batch) inside 16x16 tile
  const int kh    = lane >> 4;      // half-select per ISA VGPR layout

  // ---- one-time init: fp32 weights -> bf16 LDS (rows reordered: m = gate*4 + unit) ----
  for (int i = tid; i < 16 * K0_; i += TPB) {
    int m = i / K0_, k = i - m * K0_;
    int g = m >> 2, j = m & 3;
    w0s[i] = f2bf(W0[(size_t)(g * H_ + u0 + j) * K0_ + k]);
  }
  for (int i = tid; i < 16 * K1_; i += TPB) {
    int m = i >> 10, k = i & (K1_ - 1);
    int g = m >> 2, j = m & 3;
    w1s[i] = f2bf(W1[(size_t)(g * H_ + u0 + j) * K1_ + k]);
  }
  if (tid < 16) {
    int g = tid >> 2, j = tid & 3;
    int r = g * H_ + u0 + j;
    bias0s[tid] = b0[r] + bb0[r];
    bias1s[tid] = b1[r] + bb1[r];
  }
  // Embedding gather for t=0 (WG wg handles batch row wg); h/c regions are memset-0.
  {
    int idx = x[(size_t)wg * S_];
    for (int c = tid; c < E_; c += TPB)
      act0[(size_t)wg * K0_ + c] = f2bf(emb[(size_t)idx * E_ + c]);
  }
  float c0[2] = {0.f, 0.f}, c1[2] = {0.f, 0.f};
  gsync(sync_cnt, sync_ep);

  for (int t = 0; t < S_; ++t) {
    const int p = t & 1;

    // ================= layer 0 : z = W0slice * [xe(t), h0(t-1)]^T =================
    {
      const unsigned short* ar   = act0 + (size_t)p * B_ * K0_;
      const unsigned short* wrow = w0s + mn * K0_;
      const unsigned short* brow = ar + (size_t)(bbase + mn) * K0_;
      v8f acc = {};
      #pragma unroll 4
      for (int kb = 0; kb < K0_ / 32; ++kb) {
        const int kbase = kb * 32;
        ABu a, bm;
        a.h[0] = *(const v8bf*)(wrow + kbase + kh * 16);       // K = kh*16 + 0..7
        a.h[1] = *(const v8bf*)(wrow + kbase + kh * 16 + 8);   // K = kh*16 + 8..15
        bm.v   = *(const v16bf*)(brow + kbase + kh * 16);      // K = kh*16 + 0..15
        acc = __builtin_amdgcn_wmma_f32_16x16x32_bf16(
                  false, a.v, false, bm.v, (short)0, acc, false, false);
      }
      #pragma unroll
      for (int v = 0; v < 8; ++v) zs[wv][v + 8 * kh][mn] = acc[v];
      __syncthreads();
      #pragma unroll
      for (int e = 0; e < 2; ++e) {
        const int u = kh * 2 + e;
        const int b = bbase + mn;
        float fg = sigf    (zs[wv][ 0 + u][mn] + bias0s[ 0 + u]);
        float ig = sigf    (zs[wv][ 4 + u][mn] + bias0s[ 4 + u]);
        float ct = tanhfast(zs[wv][ 8 + u][mn] + bias0s[ 8 + u]);
        float og = sigf    (zs[wv][12 + u][mn] + bias0s[12 + u]);
        c0[e] = __builtin_fmaf(fg, c0[e], ig * ct);
        unsigned short hb = f2bf(og * tanhfast(c0[e]));
        act1[(size_t)p * B_ * K1_ + (size_t)b * K1_ + (u0 + u)] = hb;             // for layer1(t)
        act0[(size_t)(1 - p) * B_ * K0_ + (size_t)b * K0_ + (E_ + u0 + u)] = hb;  // for layer0(t+1)
      }
    }
    gsync(sync_cnt, sync_ep);

    // Overlap: gather xe(t+1) into the other act0 buffer (nobody reads it this phase).
    if (t + 1 < S_) {
      int idx = x[(size_t)wg * S_ + (t + 1)];
      for (int c = tid; c < E_; c += TPB)
        act0[(size_t)(1 - p) * B_ * K0_ + (size_t)wg * K0_ + c] =
            f2bf(emb[(size_t)idx * E_ + c]);
    }

    // ================= layer 1 : z = W1slice * [h0(t), h1(t-1)]^T =================
    {
      const unsigned short* ar   = act1 + (size_t)p * B_ * K1_;
      const unsigned short* wrow = w1s + mn * K1_;
      const unsigned short* brow = ar + (size_t)(bbase + mn) * K1_;
      v8f acc = {};
      #pragma unroll 4
      for (int kb = 0; kb < K1_ / 32; ++kb) {
        const int kbase = kb * 32;
        ABu a, bm;
        a.h[0] = *(const v8bf*)(wrow + kbase + kh * 16);
        a.h[1] = *(const v8bf*)(wrow + kbase + kh * 16 + 8);
        bm.v   = *(const v16bf*)(brow + kbase + kh * 16);
        acc = __builtin_amdgcn_wmma_f32_16x16x32_bf16(
                  false, a.v, false, bm.v, (short)0, acc, false, false);
      }
      #pragma unroll
      for (int v = 0; v < 8; ++v) zs[wv][v + 8 * kh][mn] = acc[v];
      __syncthreads();
      #pragma unroll
      for (int e = 0; e < 2; ++e) {
        const int u = kh * 2 + e;
        const int b = bbase + mn;
        float fg = sigf    (zs[wv][ 0 + u][mn] + bias1s[ 0 + u]);
        float ig = sigf    (zs[wv][ 4 + u][mn] + bias1s[ 4 + u]);
        float ct = tanhfast(zs[wv][ 8 + u][mn] + bias1s[ 8 + u]);
        float og = sigf    (zs[wv][12 + u][mn] + bias1s[12 + u]);
        c1[e] = __builtin_fmaf(fg, c1[e], ig * ct);
        unsigned short hb = f2bf(og * tanhfast(c1[e]));
        act1[(size_t)(1 - p) * B_ * K1_ + (size_t)b * K1_ + (H_ + u0 + u)] = hb; // for layer1(t+1)
      }
    }
    gsync(sync_cnt, sync_ep);
  }
}

// Final tiny FC: out[B,2] = h1_last @ fc_w.T + fc_b   (stream ordering = implicit barrier)
__global__ void fc_final(const unsigned short* __restrict__ act1,
                         const float* __restrict__ fc_w, const float* __restrict__ fc_b,
                         float* __restrict__ out)
{
  const int tid = threadIdx.x;            // 256 threads: (b,o) pairs
  const int b = tid >> 1, o = tid & 1;
  const int fbuf = 1 - ((S_ - 1) & 1);    // buffer holding h1(S-1)
  const unsigned short* hr = act1 + (size_t)fbuf * B_ * K1_ + (size_t)b * K1_ + H_;
  float s = fc_b[o];
  for (int k = 0; k < H_; ++k) s = __builtin_fmaf(bf2f(hr[k]), fc_w[o * H_ + k], s);
  out[b * 2 + o] = s;
}

extern "C" void kernel_launch(void* const* d_in, const int* in_sizes, int n_in,
                              void* d_out, int out_size, void* d_ws, size_t ws_size,
                              hipStream_t stream) {
  const int*   x    = (const int*)  d_in[0];
  const float* emb  = (const float*)d_in[1];
  const float* W0   = (const float*)d_in[2];
  const float* b0   = (const float*)d_in[3];
  const float* bb0  = (const float*)d_in[4];
  const float* W1   = (const float*)d_in[5];
  const float* b1   = (const float*)d_in[6];
  const float* bb1  = (const float*)d_in[7];
  const float* fc_w = (const float*)d_in[8];
  const float* fc_b = (const float*)d_in[9];
  float* out = (float*)d_out;

  char* ws = (char*)d_ws;
  int* sync_cnt = (int*)ws;
  int* sync_ep  = (int*)(ws + 64);
  unsigned short* act0 = (unsigned short*)(ws + 256);
  unsigned short* act1 = (unsigned short*)(ws + 256 + (size_t)2 * B_ * K0_ * sizeof(unsigned short));
  const size_t total = 256 + (size_t)2 * B_ * K0_ * 2 + (size_t)2 * B_ * K1_ * 2; // < 1 MB

  // Capture-safe reset of sync state + zero h/c activation regions each call.
  (void)hipMemsetAsync(d_ws, 0, total, stream);

  lstm_persist<<<NWG, TPB, 0, stream>>>(x, emb, W0, b0, bb0, W1, b1, bb1,
                                        sync_cnt, sync_ep, act0, act1);
  fc_final<<<1, TPB, 0, stream>>>(act1, fc_w, fc_b, out);
}
